// STGCN_75857712382315
// MI455X (gfx1250) — compile-verified, matrix operand order checked by reading, not run on hardware
//
#include <hip/hip_runtime.h>
#include <hip/hip_bf16.h>

// ---------- types for WMMA ----------
typedef __attribute__((ext_vector_type(16))) __bf16 v16bf;
typedef __attribute__((ext_vector_type(8)))  __bf16 v8bf;
typedef __attribute__((ext_vector_type(8)))  float  v8f;

#define D_IN     128
#define D_H      256
#define N_GRAPHS 128
#define N_CLS    60

static inline unsigned cdiv(unsigned a, unsigned b) { return (a + b - 1) / b; }

// ---------- utility kernels ----------
__global__ void k_zero_i32(int* __restrict__ p, int n) {
    int i = blockIdx.x * blockDim.x + threadIdx.x;
    if (i < n) p[i] = 0;
}

__global__ void k_count_edges(const int* __restrict__ edst, int* __restrict__ cnt, int E) {
    int e = blockIdx.x * blockDim.x + threadIdx.x;
    if (e < E) atomicAdd(&cnt[edst[e]], 1);
}

// dinv = rsqrt(1 + in_degree)   (the +1 is the self loop)
__global__ void k_dinv(const int* __restrict__ cnt, float* __restrict__ dinv, int n) {
    int i = blockIdx.x * blockDim.x + threadIdx.x;
    if (i < n) dinv[i] = __frsqrt_rn(1.0f + (float)cnt[i]);
}

// ---------- exclusive prefix scan of per-node counts -> rowptr ----------
__global__ void k_scan_block(const int* __restrict__ cnt, int* __restrict__ inc,
                             int* __restrict__ bsum, int n) {
    __shared__ int sm[256];
    int i = blockIdx.x * 256 + threadIdx.x;
    int v = (i < n) ? cnt[i] : 0;
    sm[threadIdx.x] = v;
    __syncthreads();
    #pragma unroll
    for (int ofs = 1; ofs < 256; ofs <<= 1) {
        int t = (threadIdx.x >= ofs) ? sm[threadIdx.x - ofs] : 0;
        __syncthreads();
        sm[threadIdx.x] += t;
        __syncthreads();
    }
    if (i < n) inc[i] = sm[threadIdx.x];
    if (threadIdx.x == 255) bsum[blockIdx.x] = sm[255];
}

__global__ void k_scan_partials(const int* __restrict__ bsum, int* __restrict__ boff,
                                int nb, int* __restrict__ rowptr, int n) {
    if (blockIdx.x != 0 || threadIdx.x != 0) return;
    int run = 0;
    for (int b = 0; b < nb; ++b) { boff[b] = run; run += bsum[b]; }
    rowptr[n] = run;      // == E
}

__global__ void k_scan_add(const int* __restrict__ inc, const int* __restrict__ cnt,
                           const int* __restrict__ boff, int* __restrict__ rowptr, int n) {
    int i = blockIdx.x * blockDim.x + threadIdx.x;
    if (i < n) rowptr[i] = inc[i] - cnt[i] + boff[i >> 8];   // exclusive scan
}

// bucket-fill CSR (dst-major adjacency holding src ids)
__global__ void k_fill_csr(const int* __restrict__ esrc, const int* __restrict__ edst,
                           const int* __restrict__ rowptr, int* __restrict__ cursor,
                           int* __restrict__ csr, int E) {
    int e = blockIdx.x * blockDim.x + threadIdx.x;
    if (e >= E) return;
    int d = edst[e];
    int pos = atomicAdd(&cursor[d], 1);
    csr[rowptr[d] + pos] = esrc[e];
}

__global__ void k_f2bf(const float* __restrict__ in, __hip_bfloat16* __restrict__ out, int n) {
    int i = blockIdx.x * blockDim.x + threadIdx.x;
    if (i < n) out[i] = __float2bfloat16(in[i]);
}

// ---------- WMMA GEMM: C[N,256] = A[N,K](bf16) * W[256,K](bf16)^T ----------
// Block = 8 waves sharing one 64-column group of W, staged into LDS via the
// async global->LDS path (ASYNCcnt). Each wave computes a 16x64 output slab:
// A fragment from global (reused 4x), B fragments from ds_load.
__global__ void __launch_bounds__(256)
k_gemm_wmma_bf16(const __hip_bfloat16* __restrict__ A,
                 const __hip_bfloat16* __restrict__ W,
                 float* __restrict__ C,
                 int N, int K) {
    __shared__ __bf16 smW[64 * D_H];                 // 32 KB max (64 rows x K)

    const int lane = threadIdx.x & 31;
    const int wave = __builtin_amdgcn_readfirstlane(threadIdx.x >> 5);
    const int tg   = blockIdx.x & 3;                 // column group (64 cols)
    const int tm   = (blockIdx.x >> 2) * 8 + wave;   // row tile for this wave
    const int nrow = N / 16;

    // ---- stage W slab [tg*64 .. tg*64+63] x K into LDS (async copy) ----
    {
        const unsigned bytes   = (unsigned)(64u * (unsigned)K * 2u);
        const unsigned ldsBase = (unsigned)(uintptr_t)(&smW[0]);
        const char*    gbase   = (const char*)(W + (size_t)tg * 64 * K);
        for (unsigned o = (unsigned)threadIdx.x * 16u; o < bytes; o += 256u * 16u) {
            unsigned long long ga = (unsigned long long)(uintptr_t)(gbase + o);
            unsigned           la = ldsBase + o;
            asm volatile("global_load_async_to_lds_b128 %0, %1, off"
                         :: "v"(la), "v"(ga) : "memory");
        }
        asm volatile("s_wait_asynccnt 0" ::: "memory");
    }
    __syncthreads();                                 // slab visible to all waves
    if (tm >= nrow) return;                          // wave-uniform: EXEC all-1

    const int arow  = tm * 16 + (lane & 15);
    const int khalf = (lane >> 4) * 8;               // 0 or 8
    const __hip_bfloat16* ap = A + (size_t)arow * K;

    v8f acc[4] = {};
    for (int k0 = 0; k0 < K; k0 += 32) {
        union { v16bf v; v8bf h[2]; } a;
        a.h[0] = *reinterpret_cast<const v8bf*>(ap + k0 + khalf);
        a.h[1] = *reinterpret_cast<const v8bf*>(ap + k0 + khalf + 16);
        #pragma unroll
        for (int t = 0; t < 4; ++t) {
            union { v16bf v; v8bf h[2]; } b;
            const int bi = (t * 16 + (lane & 15)) * K + k0 + khalf;
            b.h[0] = *reinterpret_cast<const v8bf*>(&smW[bi]);
            b.h[1] = *reinterpret_cast<const v8bf*>(&smW[bi + 16]);
            acc[t] = __builtin_amdgcn_wmma_f32_16x16x32_bf16(
                false, a.v, false, b.v, (short)0, acc[t], false, false);
        }
    }

    // D layout: VGPR r holds M = r + 8*(lane>>4), N = lane&15
    const int mbase = tm * 16 + (lane >> 4) * 8;
    #pragma unroll
    for (int t = 0; t < 4; ++t) {
        const int n = tg * 64 + t * 16 + (lane & 15);
        #pragma unroll
        for (int r = 0; r < 8; ++r)
            C[(size_t)(mbase + r) * D_H + n] = acc[t][r];
    }
}

// ---------- CSR gather + bias + relu (+ f32 out for final layer) ----------
// One wave32 per destination node; edge walk uses scalar loads (node id in SGPR).
__global__ void k_gather_bias_relu(const float* __restrict__ h,
                                   const int* __restrict__ csr,
                                   const int* __restrict__ rowptr,
                                   const float* __restrict__ dinv,
                                   const float* __restrict__ bias,
                                   __hip_bfloat16* __restrict__ nextbf,
                                   float* __restrict__ aggf32,
                                   int N, int writeF32) {
    const int lane = threadIdx.x & 31;
    const int wid  = __builtin_amdgcn_readfirstlane(
                         (int)((blockIdx.x * blockDim.x + threadIdx.x) >> 5));
    if (wid >= N) return;

    const float di = dinv[wid];
    const float4* hp = reinterpret_cast<const float4*>(h + (size_t)wid * D_H);
    float4 a0 = hp[lane], a1 = hp[lane + 32];
    const float w0 = di * di;                         // self-loop norm
    a0.x *= w0; a0.y *= w0; a0.z *= w0; a0.w *= w0;
    a1.x *= w0; a1.y *= w0; a1.z *= w0; a1.w *= w0;

    const int kb = rowptr[wid], ke = rowptr[wid + 1];
    for (int k = kb; k < ke; ++k) {
        const int s = csr[k];
        const float nrm = dinv[s] * di;
        const float4* sp = reinterpret_cast<const float4*>(h + (size_t)s * D_H);
        float4 v0 = sp[lane], v1 = sp[lane + 32];
        a0.x = fmaf(v0.x, nrm, a0.x); a0.y = fmaf(v0.y, nrm, a0.y);
        a0.z = fmaf(v0.z, nrm, a0.z); a0.w = fmaf(v0.w, nrm, a0.w);
        a1.x = fmaf(v1.x, nrm, a1.x); a1.y = fmaf(v1.y, nrm, a1.y);
        a1.z = fmaf(v1.z, nrm, a1.z); a1.w = fmaf(v1.w, nrm, a1.w);
    }

    const float4* bp = reinterpret_cast<const float4*>(bias);
    float4 b0 = bp[lane], b1 = bp[lane + 32];
    a0.x = fmaxf(a0.x + b0.x, 0.0f); a0.y = fmaxf(a0.y + b0.y, 0.0f);
    a0.z = fmaxf(a0.z + b0.z, 0.0f); a0.w = fmaxf(a0.w + b0.w, 0.0f);
    a1.x = fmaxf(a1.x + b1.x, 0.0f); a1.y = fmaxf(a1.y + b1.y, 0.0f);
    a1.z = fmaxf(a1.z + b1.z, 0.0f); a1.w = fmaxf(a1.w + b1.w, 0.0f);

    __hip_bfloat16* nb = nextbf + (size_t)wid * D_H + lane * 4;
    nb[0] = __float2bfloat16(a0.x); nb[1] = __float2bfloat16(a0.y);
    nb[2] = __float2bfloat16(a0.z); nb[3] = __float2bfloat16(a0.w);
    nb[128] = __float2bfloat16(a1.x); nb[129] = __float2bfloat16(a1.y);
    nb[130] = __float2bfloat16(a1.z); nb[131] = __float2bfloat16(a1.w);

    if (writeF32) {
        float4* ag = reinterpret_cast<float4*>(aggf32 + (size_t)wid * D_H);
        ag[lane] = a0; ag[lane + 32] = a1;
    }
}

// ---------- graph boundaries via binary search on sorted batch ----------
__global__ void k_gstart(const int* __restrict__ batch, int* __restrict__ gstart, int N) {
    int g = blockIdx.x * blockDim.x + threadIdx.x;
    if (g > N_GRAPHS) return;
    int lo = 0, hi = N;               // first i with batch[i] >= g
    while (lo < hi) { int mid = (lo + hi) >> 1; if (batch[mid] < g) lo = mid + 1; else hi = mid; }
    gstart[g] = lo;
}

// ---------- mean pool: one block per graph, thread j = column j ----------
__global__ void k_pool(const float* __restrict__ hfin, const int* __restrict__ gstart,
                       float* __restrict__ pooled) {
    const int g = blockIdx.x, j = threadIdx.x;
    const int s = gstart[g], e = gstart[g + 1];
    float sum = 0.0f;
    for (int i = s; i < e; ++i) sum += hfin[(size_t)i * D_H + j];
    pooled[(size_t)g * D_H + j] = sum / fmaxf((float)(e - s), 1.0f);
}

// ---------- FC head: out[g,c] = bfc[c] + pooledMean[g] . Wfc[c] ----------
__global__ void k_fc(const float* __restrict__ pooled, const float* __restrict__ Wfc,
                     const float* __restrict__ bfc, float* __restrict__ out, int total) {
    int i = blockIdx.x * blockDim.x + threadIdx.x;
    if (i >= total) return;
    int g = i / N_CLS, c = i % N_CLS;
    const float* p = pooled + (size_t)g * D_H;
    const float* w = Wfc + (size_t)c * D_H;
    float s = 0.0f;
    #pragma unroll 8
    for (int j = 0; j < D_H; ++j) s += p[j] * w[j];
    out[(size_t)g * N_CLS + c] = s + bfc[c];
}

// ---------- host launcher ----------
extern "C" void kernel_launch(void* const* d_in, const int* in_sizes, int n_in,
                              void* d_out, int out_size, void* d_ws, size_t ws_size,
                              hipStream_t stream) {
    const float* x    = (const float*)d_in[0];
    const int*   eidx = (const int*)  d_in[1];   // [2, E]
    const int*   batch= (const int*)  d_in[2];
    const float* W1   = (const float*)d_in[3];
    const float* b1   = (const float*)d_in[4];
    const float* W2   = (const float*)d_in[5];
    const float* b2   = (const float*)d_in[6];
    const float* W3   = (const float*)d_in[7];
    const float* b3   = (const float*)d_in[8];
    const float* Wfc  = (const float*)d_in[9];
    const float* bfc  = (const float*)d_in[10];

    const int N = in_sizes[0] / D_IN;
    const int E = in_sizes[1] / 2;
    const int* esrc = eidx;
    const int* edst = eidx + E;
    const int nb = cdiv(N, 256);

    // workspace layout (256B aligned)
    char* ws = (char*)d_ws;
    size_t off = 0;
    auto take = [&](size_t bytes) { void* p = ws + off; off = (off + bytes + 255) & ~(size_t)255; return p; };
    int*             rowcnt = (int*)            take((size_t)N * 4);
    int*             inc    = (int*)            take((size_t)N * 4);
    int*             bsum   = (int*)            take((size_t)nb * 4);
    int*             boff   = (int*)            take((size_t)nb * 4);
    int*             rowptr = (int*)            take((size_t)(N + 1) * 4);
    int*             cursor = (int*)            take((size_t)N * 4);
    int*             csr    = (int*)            take((size_t)E * 4);
    float*           dinv   = (float*)          take((size_t)N * 4);
    __hip_bfloat16*  featbf = (__hip_bfloat16*) take((size_t)N * D_H * 2);
    __hip_bfloat16*  wbf    = (__hip_bfloat16*) take((size_t)D_H * D_H * 2);
    float*           hbuf   = (float*)          take((size_t)N * D_H * 4);
    float*           agg    = (float*)          take((size_t)N * D_H * 4);
    int*             gstart = (int*)            take((size_t)(N_GRAPHS + 1) * 4);
    float*           pooled = (float*)          take((size_t)N_GRAPHS * D_H * 4);
    (void)ws_size; (void)n_in; (void)out_size;

    const int B = 256;

    // ---- degree / dinv / CSR build (once per call) ----
    k_zero_i32     <<<nb, B, 0, stream>>>(rowcnt, N);
    k_count_edges  <<<cdiv(E, B), B, 0, stream>>>(edst, rowcnt, E);
    k_dinv         <<<nb, B, 0, stream>>>(rowcnt, dinv, N);
    k_scan_block   <<<nb, B, 0, stream>>>(rowcnt, inc, bsum, N);
    k_scan_partials<<<1, 1, 0, stream>>>(bsum, boff, nb, rowptr, N);
    k_scan_add     <<<nb, B, 0, stream>>>(inc, rowcnt, boff, rowptr, N);
    k_zero_i32     <<<nb, B, 0, stream>>>(cursor, N);
    k_fill_csr     <<<cdiv(E, B), B, 0, stream>>>(esrc, edst, rowptr, cursor, csr, E);

    const int gemmGrid = 4 * cdiv(N / 16, 8);   // 4 col groups x row-tile groups
    const int gathGrid = cdiv(N, 8);            // 1 wave / node

    // ---- layer 1 (K = D_IN) ----
    k_f2bf<<<cdiv(N * D_IN, B), B, 0, stream>>>(x,  featbf, N * D_IN);
    k_f2bf<<<cdiv(D_H * D_IN, B), B, 0, stream>>>(W1, wbf,   D_H * D_IN);
    k_gemm_wmma_bf16 <<<gemmGrid, B, 0, stream>>>(featbf, wbf, hbuf, N, D_IN);
    k_gather_bias_relu<<<gathGrid, B, 0, stream>>>(hbuf, csr, rowptr, dinv, b1,
                                                   featbf, agg, N, 0);

    // ---- layer 2 (K = D_H) ----
    k_f2bf<<<cdiv(D_H * D_H, B), B, 0, stream>>>(W2, wbf, D_H * D_H);
    k_gemm_wmma_bf16 <<<gemmGrid, B, 0, stream>>>(featbf, wbf, hbuf, N, D_H);
    k_gather_bias_relu<<<gathGrid, B, 0, stream>>>(hbuf, csr, rowptr, dinv, b2,
                                                   featbf, agg, N, 0);

    // ---- layer 3 (K = D_H): keep f32 result for pooling ----
    k_f2bf<<<cdiv(D_H * D_H, B), B, 0, stream>>>(W3, wbf, D_H * D_H);
    k_gemm_wmma_bf16 <<<gemmGrid, B, 0, stream>>>(featbf, wbf, hbuf, N, D_H);
    k_gather_bias_relu<<<gathGrid, B, 0, stream>>>(hbuf, csr, rowptr, dinv, b3,
                                                   featbf, agg, N, 1);

    // ---- mean pool + FC ----
    k_gstart<<<1, 160, 0, stream>>>(batch, gstart, N);
    k_pool  <<<N_GRAPHS, D_H, 0, stream>>>(agg, gstart, pooled);
    k_fc    <<<cdiv(N_GRAPHS * N_CLS, B), B, 0, stream>>>(pooled, Wfc, bfc,
                                                          (float*)d_out, N_GRAPHS * N_CLS);
}
